// rlstm_61649960567333
// MI455X (gfx1250) — matrix operand order repaired and checked by reading, not compiled, not run on hardware
//
#include <hip/hip_runtime.h>
#include <hip/hip_bf16.h>

typedef __attribute__((ext_vector_type(16))) _Float16 v16h;
typedef __attribute__((ext_vector_type(8)))  _Float16 v8h;
typedef __attribute__((ext_vector_type(4)))  _Float16 v4h;
typedef __attribute__((ext_vector_type(8)))  float    v8f;

#define T_LEN 8192
#define NUM_CH 64
#define LSTM_H 256
#define G4 1024          // 4*H gate width
#define KTOT 320         // 64 (x) + 256 (h)
#define NUM_PROP 1024
#define NP_AFTER 256
#define PROP_LEN 128

// ---------------- workspace layout (bytes) ----------------
#define OFF_T0    ((size_t)0)                       // int[256]
#define OFF_BSUM  ((size_t)1024)                    // float[1024]
#define OFF_XWIN  ((size_t)8192)                    // f16 [256][128][64]
#define OFF_BPACK (OFF_XWIN + (size_t)256*128*64*2) // f16 WMMA-B fragments, 640KB
#define OFF_HOUT  (OFF_BPACK + (size_t)G4*KTOT*2)   // float [256][256]

__device__ __forceinline__ float fast_sigmoid(float x) {
    return 1.f / (1.f + __expf(-x));
}
__device__ __forceinline__ float fast_tanh(float x) {
    x = fminf(fmaxf(x, -15.f), 15.f);
    float e = __expf(2.f * x);
    return (e - 1.f) / (e + 1.f);
}

// ---------------- 1-D greedy NMS (one block, 1024 threads) ----------------
__global__ __launch_bounds__(1024) void nms_kernel(const float* __restrict__ props,
                                                   int* __restrict__ t0out) {
    __shared__ float ss[NUM_PROP], se[NUM_PROP];
    __shared__ int   order[NUM_PROP];
    __shared__ unsigned char supp[NUM_PROP];
    __shared__ int   totK;
    const int i = threadIdx.x;

    // rank by descending score (stable)
    float myscore = props[i * 3 + 2];
    int r = 0;
    for (int j = 0; j < NUM_PROP; ++j) {
        float oc = props[j * 3 + 2];
        r += (oc > myscore) || (oc == myscore && j < i);
    }
    order[r] = i;
    __syncthreads();
    int oi = order[i];
    ss[i] = props[oi * 3 + 0];
    se[i] = props[oi * 3 + 1];
    supp[i] = 0;
    __syncthreads();

    // greedy suppression in sorted order
    for (int p = 0; p < NUM_PROP; ++p) {
        float sp = ss[p], ep = se[p];
        int psup = supp[p];           // never written during iteration p
        if (i > p && !psup) {
            float inter = fmaxf(0.f, fminf(ep, se[i]) - fmaxf(sp, ss[i]));
            float uni   = (ep - sp) + (se[i] - ss[i]) - inter;
            float iou   = inter / fmaxf(uni, 1e-6f);
            if (iou > 0.5f) supp[i] = 1;
        }
        __syncthreads();
    }

    // compaction: kept first (sorted order), then suppressed (sorted order)
    int kb = 0;
    for (int j = 0; j < i; ++j) kb += (supp[j] == 0);
    if (i == NUM_PROP - 1) totK = kb + (supp[i] == 0);
    __syncthreads();

    int slot = -1;
    if (!supp[i]) {
        if (kb < NP_AFTER) slot = kb;
    } else {
        int s2 = totK + (i - kb);
        if (s2 < NP_AFTER) slot = s2;
    }
    if (slot >= 0) {
        int orig = order[i];
        int t0 = (int)rintf(props[orig * 3 + 0]);
        t0 = min(max(t0, 0), T_LEN - PROP_LEN);
        t0out[slot] = t0;
    }
}

// ---------------- gather windows, fp32 -> f16 ----------------
__global__ __launch_bounds__(256) void gather_kernel(const float* __restrict__ data,
                                                     const int* __restrict__ t0,
                                                     _Float16* __restrict__ Xwin) {
    int k = blockIdx.x;
    int base = t0[k];
    for (int idx = threadIdx.x; idx < PROP_LEN * NUM_CH; idx += blockDim.x) {
        int tt = idx >> 6, c = idx & 63;
        Xwin[(size_t)k * (PROP_LEN * NUM_CH) + idx] =
            (_Float16)data[(size_t)(base + tt) * NUM_CH + c];
    }
}

// ---------------- pack [W_ih | W_hh] into WMMA-B fragment order ----------------
// B is KTOT x G4; fragment (ntG, kc): 32x16 tile, B[k][n] = Wcat[n][k].
// Lane layout (16-bit B, 32x16): lanes 0-15 hold K = kc*32+0..15,
// lanes 16-31 hold K = kc*32+16..31, N = lane & 15. 16 f16 contiguous per lane.
__global__ __launch_bounds__(256) void pack_kernel(const float* __restrict__ Wih,
                                                   const float* __restrict__ Whh,
                                                   _Float16* __restrict__ Bpack) {
    int t = blockIdx.x * blockDim.x + threadIdx.x;   // (ntG*10 + kc)*32 + lane
    if (t >= (G4 / 16) * (KTOT / 32) * 32) return;
    int lane = t & 31;
    int kc   = (t >> 5) % (KTOT / 32);
    int ntG  = t / ((KTOT / 32) * 32);
    int n = ntG * 16 + (lane & 15);
    int khalf = (lane < 16) ? 0 : 16;
    _Float16* dst = Bpack + (size_t)t * 16;
    #pragma unroll
    for (int v = 0; v < 16; ++v) {
        int k = kc * 32 + khalf + v;
        float val = (k < NUM_CH) ? Wih[(size_t)n * NUM_CH + k]
                                 : Whh[(size_t)n * LSTM_H + (k - NUM_CH)];
        dst[v] = (_Float16)val;
    }
}

__global__ void bias_kernel(const float* __restrict__ bih, const float* __restrict__ bhh,
                            float* __restrict__ bsum) {
    int n = blockIdx.x * blockDim.x + threadIdx.x;
    if (n < G4) bsum[n] = bih[n] + bhh[n];
}

// ---------------- fused LSTM recurrence: 16 proposals per WG, 128 steps ----------------
__global__ __launch_bounds__(256) void lstm_kernel(const _Float16* __restrict__ Xwin,
                                                   const _Float16* __restrict__ Bpack,
                                                   const float* __restrict__ bsum,
                                                   float* __restrict__ Hout) {
    __shared__ _Float16 Alds[16][KTOT];   // [m][k]: k<64 = x(t), k>=64 = h (f16)
    __shared__ _Float16 Glds[16][G4];     // gate pre-activations (f16)

    const int tid  = threadIdx.x;
    const int lane = tid & 31;
    const int wave = tid >> 5;            // 0..7, owns gate cols [wave*128, wave*128+128)
    const int k0   = blockIdx.x * 16;     // proposal base

    // h0 = 0
    for (int idx = tid; idx < 16 * LSTM_H; idx += 256) {
        Alds[idx >> 8][NUM_CH + (idx & 255)] = (_Float16)0.f;
    }
    float c_reg[16];
    #pragma unroll
    for (int m = 0; m < 16; ++m) c_reg[m] = 0.f;

    const int rowA   = lane & 15;               // A: M = lane % 16
    const int khalfA = (lane < 16) ? 0 : 16;    // A: K-subgroup per lane half

    float biasv[8];
    #pragma unroll
    for (int nt = 0; nt < 8; ++nt)
        biasv[nt] = bsum[wave * 128 + nt * 16 + (lane & 15)];

    // per-(wave,lane) base of this wave's B fragments: 8 nt x 10 kc, 16 f16 each
    const _Float16* bbase = Bpack + ((size_t)(wave * 8) * 10 * 32 + lane) * 16;

    for (int t = 0; t < PROP_LEN; ++t) {
        // stage x(t) slice into Alds[m][0..63]
        {
            int m  = tid >> 4;
            int c4 = (tid & 15) * 4;
            const _Float16* src =
                Xwin + ((size_t)(k0 + m) * PROP_LEN + t) * NUM_CH + c4;
            *reinterpret_cast<v4h*>(&Alds[m][c4]) = *reinterpret_cast<const v4h*>(src);
            if (t + 1 < PROP_LEN)
                __builtin_prefetch(src + NUM_CH, 0, 0);   // global_prefetch_b8
        }
        __syncthreads();

        // load the 10 A fragments for this step (shared across all N tiles)
        v16h afrag[10];
        #pragma unroll
        for (int kc = 0; kc < 10; ++kc) {
            v8h lo = *reinterpret_cast<const v8h*>(&Alds[rowA][kc * 32 + khalfA]);
            v8h hi = *reinterpret_cast<const v8h*>(&Alds[rowA][kc * 32 + 8 + khalfA]);
            afrag[kc] = __builtin_shufflevector(lo, hi,
                0, 1, 2, 3, 4, 5, 6, 7, 8, 9, 10, 11, 12, 13, 14, 15);
        }

        // Defeat LICM without losing pointer provenance: launder a scalar
        // zero offset through an SGPR so the weight addresses stay
        // (global base + loop-variant scalar) -> global_load_b128 each step,
        // no hoisting, no spills.
        unsigned long long toff = 0;
        asm volatile("" : "+s"(toff));
        const _Float16* bp_t = bbase + toff;

        // gates[16 x 1024] = A[16 x 320] * B[320 x 1024] via WMMA f16->f32
        // two accumulators per N-tile to break the WMMA RAW chain
        #pragma unroll
        for (int nt = 0; nt < 8; ++nt) {
            v8f acc0 = {}, acc1 = {};
            #pragma unroll
            for (int kc = 0; kc < 10; kc += 2) {
                v16h b0 = *reinterpret_cast<const v16h*>(
                    bp_t + ((size_t)(nt * 10 + kc) * 32) * 16);
                v16h b1 = *reinterpret_cast<const v16h*>(
                    bp_t + ((size_t)(nt * 10 + kc + 1) * 32) * 16);
                acc0 = __builtin_amdgcn_wmma_f32_16x16x32_f16(
                    false, afrag[kc], false, b0, (short)0, acc0, false, false);
                acc1 = __builtin_amdgcn_wmma_f32_16x16x32_f16(
                    false, afrag[kc + 1], false, b1, (short)0, acc1, false, false);
            }
            // C/D layout: VGPR v -> M = v + (lane<16 ? 0 : 8), N = lane & 15
            int n     = wave * 128 + nt * 16 + (lane & 15);
            int mbase = (lane < 16) ? 0 : 8;
            #pragma unroll
            for (int v = 0; v < 8; ++v)
                Glds[mbase + v][n] = (_Float16)(acc0[v] + acc1[v] + biasv[nt]);
        }
        __syncthreads();

        // activation + state update: thread owns column j = tid for all 16 rows
        const int j = tid;
        const bool last = (t == PROP_LEN - 1);
        #pragma unroll
        for (int m = 0; m < 16; ++m) {
            float ig = fast_sigmoid((float)Glds[m][j]);
            float fg = fast_sigmoid((float)Glds[m][LSTM_H + j]);
            float gg = fast_tanh   ((float)Glds[m][2 * LSTM_H + j]);
            float og = fast_sigmoid((float)Glds[m][3 * LSTM_H + j]);
            float c  = fg * c_reg[m] + ig * gg;
            c_reg[m] = c;
            float h  = og * fast_tanh(c);
            Alds[m][NUM_CH + j] = (_Float16)h;
            if (last) Hout[(size_t)(k0 + m) * LSTM_H + j] = h;
        }
        __syncthreads();
    }
}

// ---------------- classification / bbox heads ----------------
__global__ __launch_bounds__(256) void head_kernel(const float* __restrict__ Hout,
                                                   const float* __restrict__ Wc,
                                                   const float* __restrict__ bc,
                                                   const float* __restrict__ Wb,
                                                   const float* __restrict__ bb,
                                                   float* __restrict__ out) {
    int t = blockIdx.x * blockDim.x + threadIdx.x;
    const int NCLS = 40, NBB = 80;
    if (t < NP_AFTER * NCLS) {
        int k = t / NCLS, n = t % NCLS;
        const float* h = Hout + (size_t)k * LSTM_H;
        const float* w = Wc + (size_t)n * LSTM_H;
        float acc = bc[n];
        for (int j = 0; j < LSTM_H; ++j) acc += h[j] * w[j];
        out[(size_t)k * NCLS + n] = acc;
    } else if (t < NP_AFTER * NCLS + NP_AFTER * NBB) {
        int t2 = t - NP_AFTER * NCLS;
        int k = t2 / NBB, n = t2 % NBB;
        const float* h = Hout + (size_t)k * LSTM_H;
        const float* w = Wb + (size_t)n * LSTM_H;
        float acc = bb[n];
        for (int j = 0; j < LSTM_H; ++j) acc += h[j] * w[j];
        out[NP_AFTER * NCLS + (size_t)k * NBB + n] = acc;
    }
}

extern "C" void kernel_launch(void* const* d_in, const int* in_sizes, int n_in,
                              void* d_out, int out_size, void* d_ws, size_t ws_size,
                              hipStream_t stream) {
    const float* data  = (const float*)d_in[0];
    const float* props = (const float*)d_in[1];
    const float* Wih   = (const float*)d_in[2];
    const float* Whh   = (const float*)d_in[3];
    const float* bih   = (const float*)d_in[4];
    const float* bhh   = (const float*)d_in[5];
    const float* Wcls  = (const float*)d_in[6];
    const float* bcls  = (const float*)d_in[7];
    const float* Wbb   = (const float*)d_in[8];
    const float* bbb   = (const float*)d_in[9];
    float* out = (float*)d_out;

    char* ws = (char*)d_ws;
    int*      t0    = (int*)(ws + OFF_T0);
    float*    bsum  = (float*)(ws + OFF_BSUM);
    _Float16* Xwin  = (_Float16*)(ws + OFF_XWIN);
    _Float16* Bpack = (_Float16*)(ws + OFF_BPACK);
    float*    Hout  = (float*)(ws + OFF_HOUT);

    nms_kernel<<<1, 1024, 0, stream>>>(props, t0);
    gather_kernel<<<NP_AFTER, 256, 0, stream>>>(data, t0, Xwin);
    pack_kernel<<<((G4 / 16) * (KTOT / 32) * 32 + 255) / 256, 256, 0, stream>>>(Wih, Whh, Bpack);
    bias_kernel<<<(G4 + 255) / 256, 256, 0, stream>>>(bih, bhh, bsum);
    lstm_kernel<<<NP_AFTER / 16, 256, 0, stream>>>(Xwin, Bpack, bsum, Hout);
    head_kernel<<<(NP_AFTER * 120 + 255) / 256, 256, 0, stream>>>(Hout, Wcls, bcls, Wbb, bbb, out);
}
